// VisionAttention_53979148976527
// MI455X (gfx1250) — matrix operand order repaired and checked by reading, not compiled, not run on hardware
//
#include <hip/hip_runtime.h>

// ---------------------------------------------------------------------------
// Types for CDNA5 WMMA (wave32): V_WMMA_F32_16X16X32_BF16
// ---------------------------------------------------------------------------
typedef __attribute__((ext_vector_type(16))) __bf16  v16bf;
typedef __attribute__((ext_vector_type(8)))  float   v8f;
typedef __attribute__((ext_vector_type(4)))  unsigned int u32x4;

union FragB16 {
    v16bf          bf;
    unsigned short us[16];
    u32x4          q[2];
};

#define WMMA_BF16(A, B, C) \
    __builtin_amdgcn_wmma_f32_16x16x32_bf16(false, (A), false, (B), (short)0, (C), false, false)

__device__ __forceinline__ unsigned short f2bf(float f) {
    unsigned int u = __float_as_uint(f);
    unsigned int r = u + 0x7FFFu + ((u >> 16) & 1u);   // round-to-nearest-even
    return (unsigned short)(r >> 16);
}
__device__ __forceinline__ float bf2f(unsigned short h) {
    return __uint_as_float(((unsigned int)h) << 16);
}

#define VZERO8 {0.f,0.f,0.f,0.f,0.f,0.f,0.f,0.f}

// ---------------------------------------------------------------------------
// Problem constants
// ---------------------------------------------------------------------------
#define SEQ     3072
#define NEMB    1280
#define NHEADS  16
#define HD      80
#define HDP     96          // head_dim padded to 3 * K32 for WMMA
#define O3      3840        // 3 * NEMB
#define SEGCAP  1024        // max segment length (cu_seqlens steps)

// ---------------------------------------------------------------------------
// fp32 -> bf16 conversion
// ---------------------------------------------------------------------------
__global__ void f32_to_bf16_kernel(const float* __restrict__ in,
                                   unsigned short* __restrict__ out, int n) {
    int i = blockIdx.x * blockDim.x + threadIdx.x;
    if (i < n) out[i] = f2bf(in[i]);
}

// ---------------------------------------------------------------------------
// TN GEMM: C[M][N] = A[M][K] (bf16) * B[N][K]^T (bf16) + bias[N]
// 8 waves / block; wave tile 32x64 (2 A frags x 4 B frags = 8 WMMA / K-step,
// 12 b128 loads / 8 WMMA). Block tile = 64x256.
// Fragment layouts as loaded here (wave32, contiguous k, same for A and B):
//   A 16x32: lanes 0-15 -> M=lane, elems 0-15 -> K=e
//            lanes 16-31 -> same M, K=16+e
//   B 32x16: lane -> N=lane&15; elems 0-15 -> K=(lane>=16?16:0)+e
//   C 16x16 f32: lane -> N=lane&15; vgpr v -> M=v+(lane>=16?8:0)
// ---------------------------------------------------------------------------
__global__ __launch_bounds__(256)
void gemm_tn_bf16_kernel(const unsigned short* __restrict__ A,
                         const unsigned short* __restrict__ B,
                         const float* __restrict__ bias,
                         void* __restrict__ Cout,
                         int M, int N, int K, int out_f32)
{
    const int lane = threadIdx.x & 31;
    const int wid  = threadIdx.x >> 5;
    const int lh   = lane & 15;
    const int hi   = lane >> 4;                 // 0 or 1 (lane half)
    const int row0 = blockIdx.x * 64  + (wid >> 2) * 32;
    const int col0 = blockIdx.y * 256 + (wid & 3) * 64;

    v8f acc[2][4];
#pragma unroll
    for (int i = 0; i < 2; ++i)
#pragma unroll
        for (int j = 0; j < 4; ++j) {
            v8f z = VZERO8;
            acc[i][j] = z;
        }

    const unsigned short* pa[2];
    const unsigned short* pb[4];
#pragma unroll
    for (int i = 0; i < 2; ++i)
        pa[i] = A + (size_t)(row0 + i * 16 + lh) * K + hi * 16;
#pragma unroll
    for (int j = 0; j < 4; ++j)
        pb[j] = B + (size_t)(col0 + j * 16 + lh) * K + hi * 16;

    for (int kb = 0; kb < K; kb += 32) {
        // prefetch next K-slab into the near cache (WGP scope)
        __builtin_prefetch((const void*)(pa[0] + kb + 128), 0, 3);
        __builtin_prefetch((const void*)(pb[0] + kb + 128), 0, 3);
        __builtin_prefetch((const void*)(pb[2] + kb + 128), 0, 3);

        FragB16 af[2], bf[4];
#pragma unroll
        for (int i = 0; i < 2; ++i) {
            const u32x4* q = (const u32x4*)(pa[i] + kb);
            af[i].q[0] = q[0]; af[i].q[1] = q[1];   // K [hi*16..+16) contiguous
        }
#pragma unroll
        for (int j = 0; j < 4; ++j) {
            const u32x4* q = (const u32x4*)(pb[j] + kb);
            bf[j].q[0] = q[0]; bf[j].q[1] = q[1];   // 32 contiguous bytes
        }
#pragma unroll
        for (int i = 0; i < 2; ++i)
#pragma unroll
            for (int j = 0; j < 4; ++j)
                acc[i][j] = WMMA_BF16(af[i].bf, bf[j].bf, acc[i][j]);
    }

#pragma unroll
    for (int i = 0; i < 2; ++i) {
#pragma unroll
        for (int j = 0; j < 4; ++j) {
            const v8f a = acc[i][j];
            const int n = col0 + j * 16 + lh;
            const float bv = bias[n];
#pragma unroll
            for (int v = 0; v < 8; ++v) {
                const int m = row0 + i * 16 + v + hi * 8;
                const float val = a[v] + bv;
                if (out_f32) ((float*)Cout)[(size_t)m * N + n] = val;
                else ((unsigned short*)Cout)[(size_t)m * N + n] = f2bf(val);
            }
        }
    }
}

// ---------------------------------------------------------------------------
// RoPE + repack: qkv[S][3*E] (bf16) -> Q/K/V [H][S][96] bf16, pad 80..95 = 0.
// Folds 1/sqrt(80) score scale into Q.
// ---------------------------------------------------------------------------
__global__ void rope_pack_kernel(const unsigned short* __restrict__ qkvb,
                                 const float* __restrict__ rpe,
                                 unsigned short* __restrict__ Qb,
                                 unsigned short* __restrict__ Kb,
                                 unsigned short* __restrict__ Vb)
{
    int idx = blockIdx.x * blockDim.x + threadIdx.x;   // over S*H*HDP
    if (idx >= SEQ * NHEADS * HDP) return;
    const int d = idx % HDP;
    const int h = (idx / HDP) % NHEADS;
    const int s = idx / (HDP * NHEADS);
    const size_t dst = ((size_t)h * SEQ + s) * HDP + d;
    if (d >= HD) { Qb[dst] = 0; Kb[dst] = 0; Vb[dst] = 0; return; }

    const unsigned short* row = qkvb + (size_t)s * O3;
    const float q = bf2f(row[h * HD + d]);
    const float k = bf2f(row[NEMB + h * HD + d]);
    const float v = bf2f(row[2 * NEMB + h * HD + d]);
    const int   dr  = (d < 40) ? d + 40 : d - 40;
    const float sgn = (d < 40) ? -1.f : 1.f;
    const float qr = sgn * bf2f(row[h * HD + dr]);
    const float kr = sgn * bf2f(row[NEMB + h * HD + dr]);
    const float ang = rpe[s * (HD / 2) + (d % 40)];
    float sn, cs;
    __sincosf(ang, &sn, &cs);
    const float scale = 0.1118033988749895f;   // 1/sqrt(80), folded into Q
    Qb[dst] = f2bf((q * cs + qr * sn) * scale);
    Kb[dst] = f2bf(k * cs + kr * sn);
    Vb[dst] = f2bf(v);
}

// ---------------------------------------------------------------------------
// Attention: one block = (head, 64 query rows). 4 waves, 128 threads.
// LDS: scores 64x1024 f32 (256KB) + reductions + transposed V tile (6KB).
// Keys restricted to the query's segment -> block-diagonal mask is free.
// ---------------------------------------------------------------------------
__global__ __launch_bounds__(128)
void attn_kernel(const unsigned short* __restrict__ Qb,
                 const unsigned short* __restrict__ Kb,
                 const unsigned short* __restrict__ Vb,
                 unsigned short* __restrict__ Ob,     // [S][E] bf16
                 const int* __restrict__ cu, int ncu)
{
    extern __shared__ char smem_raw[];
    float* s_sc  = (float*)smem_raw;                        // [64][SEGCAP]
    float* s_red = (float*)(smem_raw + 64 * SEGCAP * 4);    // [128]
    float* s_inv = s_red + 128;                             // [64]
    unsigned short* s_vt = (unsigned short*)(s_inv + 64);   // [96][32] V^T tile

    const int head  = blockIdx.y;
    const int qbase = blockIdx.x * 64;
    const int tid = threadIdx.x, lane = tid & 31, wid = tid >> 5;
    const int lh = lane & 15, hi = lane >> 4;

    int ks0 = 0, ks1 = SEQ;
    for (int i = 0; i + 1 < ncu; ++i) {
        int a = cu[i], b = cu[i + 1];
        if (qbase >= a && qbase < b) { ks0 = a; ks1 = b; }
    }
    int L = ks1 - ks0;
    if (L > SEGCAP) L = SEGCAP;

    // ---- Phase 1: scores = Q K^T (Q rows reused across all key tiles) ----
    FragB16 aq[3];
    {
        const int r = qbase + wid * 16 + lh;
        const unsigned short* qp = Qb + ((size_t)head * SEQ + r) * HDP + hi * 16;
#pragma unroll
        for (int t = 0; t < 3; ++t) {
            const u32x4* p = (const u32x4*)(qp + t * 32);
            aq[t].q[0] = p[0]; aq[t].q[1] = p[1];
        }
    }
    for (int kt = 0; kt * 16 < L; ++kt) {
        v8f acc = VZERO8;
        const int key = ks0 + kt * 16 + lh;
        const unsigned short* kp = Kb + ((size_t)head * SEQ + key) * HDP + hi * 16;
#pragma unroll
        for (int t = 0; t < 3; ++t) {
            FragB16 bk;
            const u32x4* p = (const u32x4*)(kp + t * 32);
            bk.q[0] = p[0]; bk.q[1] = p[1];
            acc = WMMA_BF16(aq[t].bf, bk.bf, acc);
        }
#pragma unroll
        for (int v = 0; v < 8; ++v)
            s_sc[(wid * 16 + v + hi * 8) * SEGCAP + kt * 16 + lh] = acc[v];
    }
    __syncthreads();

    // ---- Phase 2: two-pass softmax over LDS rows ----
    {
        const int row = tid & 63, half = tid >> 6, c0 = half * (SEGCAP / 2);
        float m = -3.4e38f;
        for (int c = c0; c < c0 + SEGCAP / 2; ++c)
            if (c < L) m = fmaxf(m, s_sc[row * SEGCAP + c]);
        s_red[half * 64 + row] = m;
        __syncthreads();
        m = fmaxf(s_red[row], s_red[64 + row]);
        __syncthreads();
        float sum = 0.f;
        for (int c = c0; c < c0 + SEGCAP / 2; ++c) {
            if (c < L) {
                float e = __expf(s_sc[row * SEGCAP + c] - m);
                s_sc[row * SEGCAP + c] = e;
                sum += e;
            }
        }
        s_red[half * 64 + row] = sum;
        __syncthreads();
        if (half == 0) {
            float tot = s_red[row] + s_red[64 + row];
            s_inv[row] = (tot > 0.f) ? 1.f / tot : 0.f;
        }
        __syncthreads();
    }

    // ---- Phase 3: O = P V, V tile staged transposed into LDS ----
    v8f o[5] = {VZERO8, VZERO8, VZERO8, VZERO8, VZERO8};
    for (int kk = 0; kk < L; kk += 32) {
        // cooperative coalesced load of V[kk..kk+31][0..95] -> s_vt[hd][key]
        for (int idx = tid; idx < 32 * (HDP / 2); idx += 128) {
            const int key = idx / (HDP / 2);
            const int cu32 = idx % (HDP / 2);
            const unsigned int u =
                ((const unsigned int*)Vb)[((size_t)head * SEQ + ks0 + kk + key) * (HDP / 2) + cu32];
            s_vt[(2 * cu32) * 32 + key]     = (unsigned short)(u & 0xFFFFu);
            s_vt[(2 * cu32 + 1) * 32 + key] = (unsigned short)(u >> 16);
        }
        __syncthreads();

        // P fragment (A 16x32), 1/rowsum folded in
        FragB16 pa;
        {
            const int rl = wid * 16 + lh;
            const float inv = s_inv[rl];
#pragma unroll
            for (int e = 0; e < 16; ++e) {
                const int koff = hi * 16 + e; // contiguous k per lane half
                pa.us[e] = f2bf(s_sc[rl * SEGCAP + kk + koff] * inv);
            }
        }
#pragma unroll
        for (int nt = 0; nt < 5; ++nt) {
            FragB16 bv;
            const u32x4* p = (const u32x4*)(s_vt + (nt * 16 + lh) * 32 + hi * 16);
            bv.q[0] = p[0]; bv.q[1] = p[1];
            o[nt] = WMMA_BF16(pa.bf, bv.bf, o[nt]);
        }
        __syncthreads();
    }

    // ---- writeback: attn_out[S][E] bf16 (already normalized) ----
#pragma unroll
    for (int nt = 0; nt < 5; ++nt) {
        const int col = head * HD + nt * 16 + lh;
#pragma unroll
        for (int v = 0; v < 8; ++v) {
            const int row = qbase + wid * 16 + v + hi * 8;
            Ob[(size_t)row * NEMB + col] = f2bf(o[nt][v]);
        }
    }
}

// ---------------------------------------------------------------------------
// Host side
// ---------------------------------------------------------------------------
extern "C" void kernel_launch(void* const* d_in, const int* in_sizes, int n_in,
                              void* d_out, int out_size, void* d_ws, size_t ws_size,
                              hipStream_t stream) {
    const float* x     = (const float*)d_in[0];
    const float* Wqkv  = (const float*)d_in[1];
    const float* bqkv  = (const float*)d_in[2];
    const float* Wproj = (const float*)d_in[3];
    const float* bproj = (const float*)d_in[4];
    const float* rpe   = (const float*)d_in[5];
    const int*   cu    = (const int*)d_in[6];
    const int    ncu   = in_sizes[6];
    float* out = (float*)d_out;

    // workspace layout (256B aligned)
    char* ws = (char*)d_ws;
    size_t off = 0;
    auto take = [&](size_t bytes) -> void* {
        void* p = (void*)(ws + off);
        off += (bytes + 255) & ~(size_t)255;
        return p;
    };
    unsigned short* xb     = (unsigned short*)take((size_t)SEQ * NEMB * 2);
    unsigned short* Wqkvb  = (unsigned short*)take((size_t)O3 * NEMB * 2);
    unsigned short* Wprojb = (unsigned short*)take((size_t)NEMB * NEMB * 2);
    unsigned short* qkvb   = (unsigned short*)take((size_t)SEQ * O3 * 2);
    unsigned short* Qb     = (unsigned short*)take((size_t)NHEADS * SEQ * HDP * 2);
    unsigned short* Kb     = (unsigned short*)take((size_t)NHEADS * SEQ * HDP * 2);
    unsigned short* Vb     = (unsigned short*)take((size_t)NHEADS * SEQ * HDP * 2);
    unsigned short* Ob     = (unsigned short*)take((size_t)SEQ * NEMB * 2);

    // 1) convert inputs to bf16 (everything stays L2-resident afterwards)
    {
        int n1 = SEQ * NEMB, n2 = O3 * NEMB, n3 = NEMB * NEMB;
        f32_to_bf16_kernel<<<(n1 + 255) / 256, 256, 0, stream>>>(x, xb, n1);
        f32_to_bf16_kernel<<<(n2 + 255) / 256, 256, 0, stream>>>(Wqkv, Wqkvb, n2);
        f32_to_bf16_kernel<<<(n3 + 255) / 256, 256, 0, stream>>>(Wproj, Wprojb, n3);
    }

    // 2) qkv = x @ W_qkv^T + b_qkv   (bf16 out)
    gemm_tn_bf16_kernel<<<dim3(SEQ / 64, O3 / 256), 256, 0, stream>>>(
        xb, Wqkvb, bqkv, (void*)qkvb, SEQ, O3, NEMB, /*out_f32=*/0);

    // 3) RoPE + repack to [H][S][96] with zero padding, fold 1/sqrt(hd) into Q
    {
        int n = SEQ * NHEADS * HDP;
        rope_pack_kernel<<<(n + 255) / 256, 256, 0, stream>>>(qkvb, rpe, Qb, Kb, Vb);
    }

    // 4) block-diagonal attention (scores in 256KB LDS, softmax, PV)
    {
        size_t lds = (size_t)64 * SEGCAP * 4 + 128 * 4 + 64 * 4 + (size_t)HDP * 32 * 2;
        attn_kernel<<<dim3(SEQ / 64, NHEADS), 128, lds, stream>>>(Qb, Kb, Vb, Ob, cu, ncu);
    }

    // 5) out = attn_out @ W_proj^T + b_proj   (fp32 out)
    gemm_tn_bf16_kernel<<<dim3(SEQ / 64, NEMB / 256), 256, 0, stream>>>(
        Ob, Wprojb, bproj, (void*)out, SEQ, NEMB, NEMB, /*out_f32=*/1);
}
